// CausalSelfAttention_63402307223804
// MI455X (gfx1250) — matrix operand order repaired and checked by reading, not compiled, not run on hardware
//
#include <hip/hip_runtime.h>

#define BB 4
#define TT 2048
#define CC 768
#define HH 12
#define DD 64
#define BH (BB * HH)

typedef __attribute__((ext_vector_type(16))) __bf16 v16bf;
typedef __attribute__((ext_vector_type(8)))  float  v8f;

// ---------------------------------------------------------------------------
// Fragment loaders per CDNA5 ISA 7.12.2 (wave32, 16-bit operands, 16x16x32)
// A (16x32, MxK): lane -> M = lane&15 ; kg = lane>>4
//   elems 0..7  : K = kg*8 + j ; elems 8..15 : K = 16 + kg*8 + (j-8)
// B (32x16, KxN): lane -> N = lane&15 ; lanes 0-15 hold K=0..7,16..23; 16-31 K=8..15,24..31
// C/D (16x16 f32): lane -> N = lane&15 ; VGPR r -> M = r + 8*(lane>>4)
// All B-operand sources are staged K-major in LDS so every fragment load is
// two runs of 8 contiguous bf16 (2 x ds_load_b128), no scalar column gathers.
// ---------------------------------------------------------------------------
__device__ __forceinline__ v16bf load_A_lds(const __bf16* p, int ld, int row0, int k0) {
  const int lane = threadIdx.x & 31;
  const __bf16* rp = p + (row0 + (lane & 15)) * ld + k0 + ((lane >> 4) << 3);
  v16bf a;
#pragma unroll
  for (int j = 0; j < 8; ++j) a[j] = rp[j];
#pragma unroll
  for (int j = 0; j < 8; ++j) a[8 + j] = rp[16 + j];
  return a;
}

// B[k][n] where LDS storage is [n][k] (K-major): contiguous per lane.
__device__ __forceinline__ v16bf load_B_ldsT(const __bf16* p, int ld, int n0, int k0) {
  const int lane = threadIdx.x & 31;
  const __bf16* rp = p + (n0 + (lane & 15)) * ld + k0 + ((lane >> 4) << 3);
  v16bf b;
#pragma unroll
  for (int j = 0; j < 8; ++j) b[j] = rp[j];
#pragma unroll
  for (int j = 0; j < 8; ++j) b[8 + j] = rp[16 + j];
  return b;
}

// ---------------------------------------------------------------------------
// Tiled GEMM + bias. 64x64 output tile per block (128 thr = 4 wave32).
// Wave w owns columns [w*16, w*16+16) x all 64 rows: 4 wmma accumulators.
// W tile is transposed into LDS (Wt[n][k]) so B fragments are contiguous.
// MODE 0: A fp32 (x); epilogue scatters bf16 into q/k [B,H,T,D], v [B,H,D,T].
// MODE 1: A bf16 (y); epilogue writes fp32 + bias to output.
// ---------------------------------------------------------------------------
template <int MODE>
__global__ __launch_bounds__(128) void gemm_bias(const void* __restrict__ Ap,
                                                 const float* __restrict__ W,
                                                 const float* __restrict__ bias,
                                                 __bf16* __restrict__ qo,
                                                 __bf16* __restrict__ ko,
                                                 __bf16* __restrict__ vo,
                                                 float* __restrict__ outf,
                                                 int Mtot, int Ntot, int Ktot) {
  __shared__ alignas(16) __bf16 As[64 * 40];   // 64 rows x 32 k (+pad)
  __shared__ alignas(16) __bf16 Wt[64 * 40];   // 64 n x 32 k (+pad), transposed
  const int tid  = threadIdx.x;
  const int lane = tid & 31;
  const int wave = tid >> 5;
  const int tileN = blockIdx.x * 64;
  const int tileM = blockIdx.y * 64;

  v8f acc[4];
#pragma unroll
  for (int g = 0; g < 4; ++g)
#pragma unroll
    for (int r = 0; r < 8; ++r) acc[g][r] = 0.0f;

  for (int k0 = 0; k0 < Ktot; k0 += 32) {
    __syncthreads();
    if (MODE == 0) {
      const float* A = (const float*)Ap;
#pragma unroll
      for (int i = 0; i < 4; ++i) {
        int idx = tid + i * 128;               // 512 float4 = 64x32 tile
        int row = idx >> 3;
        int col = (idx & 7) << 2;
        const float4 f4 = *(const float4*)(A + (size_t)(tileM + row) * Ktot + k0 + col);
        __bf16* d = As + row * 40 + col;
        d[0] = (__bf16)f4.x; d[1] = (__bf16)f4.y; d[2] = (__bf16)f4.z; d[3] = (__bf16)f4.w;
      }
    } else {
      const __bf16* A = (const __bf16*)Ap;
#pragma unroll
      for (int i = 0; i < 16; ++i) {
        int idx = tid + i * 128;
        int row = idx >> 5;
        int col = idx & 31;
        As[row * 40 + col] = A[(size_t)(tileM + row) * Ktot + k0 + col];
      }
    }
    // W tile [32 k][64 n] fp32 -> transposed bf16 Wt[n][k]
#pragma unroll
    for (int i = 0; i < 4; ++i) {
      int idx = tid + i * 128;                 // 512 float4 = 32x64 tile
      int kk  = idx >> 4;                      // k row 0..31
      int nn  = (idx & 15) << 2;               // n col 0..60
      const float4 f4 = *(const float4*)(W + (size_t)(k0 + kk) * Ntot + tileN + nn);
      Wt[(nn + 0) * 40 + kk] = (__bf16)f4.x;
      Wt[(nn + 1) * 40 + kk] = (__bf16)f4.y;
      Wt[(nn + 2) * 40 + kk] = (__bf16)f4.z;
      Wt[(nn + 3) * 40 + kk] = (__bf16)f4.w;
    }
    __syncthreads();

    const v16bf bf = load_B_ldsT(Wt, 40, wave * 16, 0);
#pragma unroll
    for (int g = 0; g < 4; ++g) {
      const v16bf af = load_A_lds(As, 40, g * 16, 0);
      acc[g] = __builtin_amdgcn_wmma_f32_16x16x32_bf16(false, af, false, bf, (short)0,
                                                       acc[g], false, false);
    }
  }

#pragma unroll
  for (int g = 0; g < 4; ++g) {
#pragma unroll
    for (int r = 0; r < 8; ++r) {
      const int gm = tileM + g * 16 + r + ((lane >> 4) << 3);
      const int gn = tileN + wave * 16 + (lane & 15);
      const float val = acc[g][r] + bias[gn];
      if (MODE == 0) {
        const int b = gm >> 11;                // / T
        const int t = gm & (TT - 1);
        const int head3 = gn >> 6;             // / D, 0..35
        const int d = gn & (DD - 1);
        const int which = head3 / HH;          // 0=q 1=k 2=v
        const int h = head3 - which * HH;
        if (which == 2) {                      // V stored [B,H,D,T] (K-major)
          vo[(((size_t)(b * HH + h) * DD) + d) * TT + t] = (__bf16)val;
        } else {
          __bf16* dst = (which == 0) ? qo : ko;
          dst[(((size_t)(b * HH + h) * TT) + t) * DD + d] = (__bf16)val;
        }
      } else {
        outf[(size_t)gm * Ntot + gn] = val;
      }
    }
  }
}

// ---------------------------------------------------------------------------
// Flash attention: one block per (64-query tile, b*h). 128 thr = 4 waves,
// wave w owns queries [w*16, w*16+16). Online softmax over 64-key blocks.
// K stored [bh][t][d] (row = key, contiguous d -> K^T fragment contiguous),
// V stored [bh][d][t] (row = d, contiguous key -> PV fragment contiguous).
// ---------------------------------------------------------------------------
__global__ __launch_bounds__(128) void flash_attn(const __bf16* __restrict__ q,
                                                  const __bf16* __restrict__ k,
                                                  const __bf16* __restrict__ v,
                                                  __bf16* __restrict__ y) {
  __shared__ alignas(16) __bf16 Qs[64 * 64];
  __shared__ alignas(16) __bf16 Ks[64 * 64];   // [key][d]
  __shared__ alignas(16) __bf16 Vt[64 * 64];   // [d][key]
  __shared__ alignas(16) __bf16 Ps[4 * 16 * 64];  // wave-private P staging
  const int tid   = threadIdx.x;
  const int lane  = tid & 31;
  const int wave  = tid >> 5;
  const int qtile = blockIdx.x;
  const int bh    = blockIdx.y;

  {
    const uint4* src = (const uint4*)(q + ((size_t)bh * TT + qtile * 64) * DD);
    uint4* dst = (uint4*)Qs;
#pragma unroll
    for (int i = 0; i < 4; ++i) dst[tid + i * 128] = src[tid + i * 128];
  }
  __syncthreads();
  const v16bf qA0 = load_A_lds(Qs, 64, wave * 16, 0);
  const v16bf qA1 = load_A_lds(Qs, 64, wave * 16, 32);

  float m_run[8], l_run[8];
  v8f o[4];
#pragma unroll
  for (int r = 0; r < 8; ++r) { m_run[r] = -3.0e38f; l_run[r] = 0.0f; }
#pragma unroll
  for (int f = 0; f < 4; ++f)
#pragma unroll
    for (int r = 0; r < 8; ++r) o[f][r] = 0.0f;

  for (int j = 0; j <= qtile; ++j) {           // causal: only lower blocks
    __syncthreads();
    {
      // K tile: fully contiguous 64x64 block
      const uint4* ks = (const uint4*)(k + ((size_t)bh * TT + j * 64) * DD);
      uint4* kd = (uint4*)Ks;
      // V tile: 64 rows (d) of 64 keys, row stride T in global
      const uint4* vs = (const uint4*)(v + (size_t)bh * DD * TT + j * 64);
      uint4* vd = (uint4*)Vt;
#pragma unroll
      for (int i = 0; i < 4; ++i) {
        int idx = tid + i * 128;
        kd[idx] = ks[idx];
        int dr = idx >> 3;                     // d row
        int cc = idx & 7;                      // uint4 col within 64 keys
        vd[dr * 8 + cc] = vs[dr * (TT / 8) + cc];
      }
    }
    __syncthreads();

    // S = Q K^T : 4 frags of 16q x 16key, contracted over d in 2 chunks of 32
    v8f s[4];
#pragma unroll
    for (int f = 0; f < 4; ++f) {
      v8f z = {0.f, 0.f, 0.f, 0.f, 0.f, 0.f, 0.f, 0.f};
      z = __builtin_amdgcn_wmma_f32_16x16x32_bf16(false, qA0, false,
            load_B_ldsT(Ks, 64, f * 16, 0), (short)0, z, false, false);
      z = __builtin_amdgcn_wmma_f32_16x16x32_bf16(false, qA1, false,
            load_B_ldsT(Ks, 64, f * 16, 32), (short)0, z, false, false);
      s[f] = z;
    }

    const bool diag = (j == qtile);
    const int m_lo = (lane >> 4) << 3;
#pragma unroll
    for (int f = 0; f < 4; ++f)
#pragma unroll
      for (int r = 0; r < 8; ++r) {
        float val = s[f][r] * 0.125f;          // 1/sqrt(64)
        if (diag) {
          const int key = f * 16 + (lane & 15);
          const int qr  = wave * 16 + m_lo + r;
          if (key > qr) val = -3.0e38f;
        }
        s[f][r] = val;
      }

    // row max across the 16-lane half owning each fragment column set
    float bm[8];
#pragma unroll
    for (int r = 0; r < 8; ++r)
      bm[r] = fmaxf(fmaxf(s[0][r], s[1][r]), fmaxf(s[2][r], s[3][r]));
#pragma unroll
    for (int sh = 1; sh <= 8; sh <<= 1)
#pragma unroll
      for (int r = 0; r < 8; ++r) bm[r] = fmaxf(bm[r], __shfl_xor(bm[r], sh, 32));

    float alpha[8], rs[8];
#pragma unroll
    for (int r = 0; r < 8; ++r) {
      const float mn = fmaxf(m_run[r], bm[r]);
      alpha[r] = __expf(m_run[r] - mn);
      m_run[r] = mn;
      rs[r] = 0.0f;
    }

    // P = exp(S - m), stage to wave-private LDS (C-layout -> A-layout)
#pragma unroll
    for (int f = 0; f < 4; ++f)
#pragma unroll
      for (int r = 0; r < 8; ++r) {
        const float pv = __expf(s[f][r] - m_run[r]);
        rs[r] += pv;
        Ps[wave * 1024 + (m_lo + r) * 64 + f * 16 + (lane & 15)] = (__bf16)pv;
      }
#pragma unroll
    for (int sh = 1; sh <= 8; sh <<= 1)
#pragma unroll
      for (int r = 0; r < 8; ++r) rs[r] += __shfl_xor(rs[r], sh, 32);
#pragma unroll
    for (int r = 0; r < 8; ++r) l_run[r] = l_run[r] * alpha[r] + rs[r];
#pragma unroll
    for (int f = 0; f < 4; ++f)
#pragma unroll
      for (int r = 0; r < 8; ++r) o[f][r] *= alpha[r];

    // O += P V : contract over 64 keys in 2 chunks of 32; V is K-major in LDS
    const v16bf pA0 = load_A_lds(Ps + wave * 1024, 64, 0, 0);
    const v16bf pA1 = load_A_lds(Ps + wave * 1024, 64, 0, 32);
#pragma unroll
    for (int g = 0; g < 4; ++g) {
      o[g] = __builtin_amdgcn_wmma_f32_16x16x32_bf16(false, pA0, false,
               load_B_ldsT(Vt, 64, g * 16, 0), (short)0, o[g], false, false);
      o[g] = __builtin_amdgcn_wmma_f32_16x16x32_bf16(false, pA1, false,
               load_B_ldsT(Vt, 64, g * 16, 32), (short)0, o[g], false, false);
    }
  }

  // normalize and write y as bf16 in [B,T,C] for the projection GEMM
  const int b = bh / HH;
  const int h = bh - b * HH;
#pragma unroll
  for (int f = 0; f < 4; ++f)
#pragma unroll
    for (int r = 0; r < 8; ++r) {
      const int m = r + ((lane >> 4) << 3);
      const int t = qtile * 64 + wave * 16 + m;
      const int d = f * 16 + (lane & 15);
      const float val = o[f][r] / l_run[r];
      y[((size_t)(b * TT + t)) * CC + h * DD + d] = (__bf16)val;
    }
}

// ---------------------------------------------------------------------------
extern "C" void kernel_launch(void* const* d_in, const int* in_sizes, int n_in,
                              void* d_out, int out_size, void* d_ws, size_t ws_size,
                              hipStream_t stream) {
  const float* x      = (const float*)d_in[0];
  // d_in[1] = causal mask (bool) -- causality derived analytically in-kernel
  const float* W_attn = (const float*)d_in[2];
  const float* b_attn = (const float*)d_in[3];
  const float* W_proj = (const float*)d_in[4];
  const float* b_proj = (const float*)d_in[5];
  float* out = (float*)d_out;

  const size_t elems = (size_t)BH * TT * DD;   // 6291456 (== B*T*C as well)
  char* wsb = (char*)d_ws;
  __bf16* qb = (__bf16*)(wsb);
  __bf16* kb = (__bf16*)(wsb + 2 * elems);
  __bf16* vb = (__bf16*)(wsb + 4 * elems);     // [B,H,D,T]
  __bf16* yb = (__bf16*)(wsb + 6 * elems);

  gemm_bias<0><<<dim3((3 * CC) / 64, (BB * TT) / 64), 128, 0, stream>>>(
      x, W_attn, b_attn, qb, kb, vb, nullptr, BB * TT, 3 * CC, CC);
  flash_attn<<<dim3(TT / 64, BH), 128, 0, stream>>>(qb, kb, vb, yb);
  gemm_bias<1><<<dim3(CC / 64, (BB * TT) / 64), 128, 0, stream>>>(
      yb, W_proj, b_proj, nullptr, nullptr, nullptr, out, BB * TT, CC, CC);
}
